// RepFlowLayer_33088428049224
// MI455X (gfx1250) — matrix operand order repaired and checked by reading, not compiled, not run on hardware
//
#include <hip/hip_runtime.h>

// ---------------------------------------------------------------------------
// MI455X (gfx1250) RepFlow layer.
// All GEMMs run on v_wmma_f32_16x16x32_bf16 (wave32, 8 waves / 256-thread
// block). A-tiles are staged in LDS via GLOBAL_LOAD_ASYNC_TO_LDS_B128 when
// the builtin is available (ASYNCcnt + s_wait_asynccnt), else via b128
// copies. Weights are pre-packed into the WMMA B-fragment lane layout and
// stay hot in the 192MB L2. f32 accumulate, bf16 operands.
// ---------------------------------------------------------------------------

typedef __bf16 bhalf;
typedef __attribute__((ext_vector_type(16))) __bf16 v16bf;
typedef __attribute__((ext_vector_type(8)))  float  v8f;

union FragU { v16bf v; uint4 q[2]; };

__device__ __forceinline__ v8f wmma_bf16(v16bf a, v16bf b, v8f c) {
    return __builtin_amdgcn_wmma_f32_16x16x32_bf16(
        /*neg_a=*/false, a, /*neg_b=*/false, b,
        /*c_mod=*/(short)0, c, /*reuse_a=*/false, /*reuse_b=*/false);
}

__device__ __forceinline__ float silu_f(float x) {
    return x / (1.0f + __expf(-x));
}

__device__ __forceinline__ v8f zero8() {
    v8f z = {0.f,0.f,0.f,0.f,0.f,0.f,0.f,0.f};
    return z;
}

// ---- async global->LDS staging (CDNA5), guarded so both passes compile ----
#if __has_builtin(__builtin_amdgcn_global_load_async_to_lds_b128)
#define HAVE_ASYNC_LDS 1
typedef int v4i_vs __attribute__((vector_size(16)));
typedef __attribute__((address_space(1))) v4i_vs* gptr_b128;
typedef __attribute__((address_space(3))) v4i_vs* lptr_b128;
__device__ __forceinline__ void cp16(const bhalf* g, bhalf* l) {
    __builtin_amdgcn_global_load_async_to_lds_b128(
        (gptr_b128)g, (lptr_b128)l, 0, 0);
}
__device__ __forceinline__ void cp_wait() {
    asm volatile("s_wait_asynccnt 0x0" ::: "memory");
}
#else
#define HAVE_ASYNC_LDS 0
__device__ __forceinline__ void cp16(const bhalf* g, bhalf* l) {
    *(uint4*)l = *(const uint4*)g;
}
__device__ __forceinline__ void cp_wait() {}
#endif

// A fragment: lane L in [0,16): M = mbase+L, K-offsets [g*8, g*8+8) and
// [16+g*8, ...) with g = L>>4. Two 16B loads per lane.
__device__ __forceinline__ v16bf load_a_frag(const bhalf* base, int stride,
                                             int mbase, int kbase, int lane) {
    int m = mbase + (lane & 15);
    int g = lane >> 4;
    const bhalf* p = base + m * stride + kbase + g * 8;
    FragU f;
    f.q[0] = *(const uint4*)(p);
    f.q[1] = *(const uint4*)(p + 16);
    return f.v;
}

// B fragment: pre-packed 1KB blobs, 32 lanes x 32B contiguous.
__device__ __forceinline__ v16bf load_b_frag(const uint4* pk, int kb, int nt,
                                             int NT, int lane) {
    const uint4* p = pk + ((kb * NT + nt) * 32 + lane) * 2;
    FragU f;
    f.q[0] = p[0];
    f.q[1] = p[1];
    return f.v;
}

__device__ __forceinline__ unsigned bf16bits(float x) {
    unsigned u = __float_as_uint(x);
    unsigned r = (u + 0x7FFFu + ((u >> 16) & 1u)) >> 16;
    return r & 0xFFFFu;
}

// --------------------------- elementwise prep ------------------------------

__global__ void k_cvt(const float* __restrict__ src, bhalf* __restrict__ dst, int n) {
    for (int i = blockIdx.x * blockDim.x + threadIdx.x; i < n; i += gridDim.x * blockDim.x)
        dst[i] = (bhalf)src[i];
}

// ea = edge_ebd[:, :16] * a_nlist_mask  -> bf16 [1024,16,64]
__global__ void k_ea(const float* __restrict__ edge, const unsigned char* __restrict__ amask,
                     bhalf* __restrict__ ea) {
    int n = 1024 * 16 * 64;
    for (int idx = blockIdx.x * blockDim.x + threadIdx.x; idx < n; idx += gridDim.x * blockDim.x) {
        int c = idx & 63;
        int a = (idx >> 6) & 15;
        int loc = idx >> 10;
        float m = amask[loc * 16 + a] ? 1.0f : 0.0f;
        ea[idx] = (bhalf)(edge[loc * 4096 + a * 64 + c] * m);
    }
}

// Pack (optionally fused pair of) row-major f32 weights [K, n0 (+ n1)] into
// WMMA B fragments: k = kb*32 + (lane>>4)*8 + (v&3)*2 + (v>>2)*16, n = nt*16 + (lane&15).
__global__ void k_pack(const float* __restrict__ W0, int n0,
                       const float* __restrict__ W1, int n1,
                       int K, unsigned* __restrict__ dst) {
    int N = n0 + n1;
    int total = (K / 32) * (N / 16) * 32 * 8;
    for (int idx = blockIdx.x * blockDim.x + threadIdx.x; idx < total; idx += gridDim.x * blockDim.x) {
        int v = idx & 7;
        int lane = (idx >> 3) & 31;
        int frag = idx >> 8;
        int nt = frag % (N / 16);
        int kb = frag / (N / 16);
        int n = nt * 16 + (lane & 15);
        int k = kb * 32 + (lane >> 4) * 8 + (v & 3) * 2 + (v >> 2) * 16;
        float x0, x1;
        if (n < n0) { x0 = W0[k * n0 + n];        x1 = W0[(k + 1) * n0 + n]; }
        else        { x0 = W1[k * n1 + (n - n0)]; x1 = W1[(k + 1) * n1 + (n - n0)]; }
        dst[idx] = bf16bits(x0) | (bf16bits(x1) << 16);
    }
}

// --------------------------- sym (VALU, tiny) ------------------------------
// Per loc: h2g2 = h2^T(3x64) . (ebd*msw), grrg[a][d] = sum_c h2g2[c][a]*h2g2[c][d]/3
__global__ void __launch_bounds__(128)
k_sym(const float* __restrict__ node_f, const float* __restrict__ edge_f,
      const float* __restrict__ h2, const int* __restrict__ nlist,
      const unsigned char* __restrict__ nmask, const float* __restrict__ sw,
      bhalf* __restrict__ sym) {
    int loc = blockIdx.x;
    int t = threadIdx.x;
    __shared__ float sh2[64][4];
    __shared__ float smsw[64];
    __shared__ int   snl[64];
    __shared__ float se[3][64];
    __shared__ float sn[3][128];
    if (t < 64) {
        smsw[t] = (nmask[loc * 64 + t] ? 1.0f : 0.0f) * sw[loc * 64 + t];
        snl[t] = nlist[loc * 64 + t];
        sh2[t][0] = h2[(loc * 64 + t) * 3 + 0];
        sh2[t][1] = h2[(loc * 64 + t) * 3 + 1];
        sh2[t][2] = h2[(loc * 64 + t) * 3 + 2];
    }
    __syncthreads();
    if (t < 64) {
        float a0 = 0.f, a1 = 0.f, a2 = 0.f;
        for (int nn = 0; nn < 64; nn++) {
            float g = edge_f[loc * 4096 + nn * 64 + t] * smsw[nn];
            a0 += sh2[nn][0] * g; a1 += sh2[nn][1] * g; a2 += sh2[nn][2] * g;
        }
        se[0][t] = a0 * 0.125f; se[1][t] = a1 * 0.125f; se[2][t] = a2 * 0.125f;
    }
    {
        float a0 = 0.f, a1 = 0.f, a2 = 0.f;
        for (int nn = 0; nn < 64; nn++) {
            float g = node_f[snl[nn] * 128 + t] * smsw[nn];
            a0 += sh2[nn][0] * g; a1 += sh2[nn][1] * g; a2 += sh2[nn][2] * g;
        }
        sn[0][t] = a0 * 0.125f; sn[1][t] = a1 * 0.125f; sn[2][t] = a2 * 0.125f;
    }
    __syncthreads();
    for (int e = t; e < 768; e += 128) {
        float v;
        if (e < 256) {
            int a = e >> 6, d = e & 63;
            v = (se[0][a] * se[0][d] + se[1][a] * se[1][d] + se[2][a] * se[2][d]) * (1.0f / 3.0f);
        } else {
            int e2 = e - 256;
            int a = e2 >> 7, d = e2 & 127;
            v = (sn[0][a] * sn[0][d] + sn[1][a] * sn[1][d] + sn[2][a] * sn[2][d]) * (1.0f / 3.0f);
        }
        sym[loc * 768 + e] = (bhalf)v;
    }
}

// --------------------------- node GEMMs -----------------------------------
// n_partial = node + res_n0*silu(node@Wns+b) + res_n1*silu(sym@Wsym+b)
__global__ void __launch_bounds__(256)
k_node(const float* __restrict__ node_f, const bhalf* __restrict__ node_bf,
       const bhalf* __restrict__ sym_bf, const uint4* __restrict__ pk_ns,
       const uint4* __restrict__ pk_sym, const float* __restrict__ b_ns,
       const float* __restrict__ b_sym, const float* __restrict__ res_n0,
       const float* __restrict__ res_n1, float* __restrict__ n_partial) {
    int rowbase = blockIdx.x * 64;
    int lane = threadIdx.x & 31, w = threadIdx.x >> 5;
    int mt = w & 3, ntb = (w >> 2) * 4;
    const bhalf* A1 = node_bf + rowbase * 128;
    const bhalf* A2 = sym_bf + rowbase * 768;
    for (int nti = 0; nti < 4; nti++) {
        int nt = ntb + nti;
        v8f acc1 = zero8();
        for (int kb = 0; kb < 4; kb++) {
            v16bf a = load_a_frag(A1, 128, mt * 16, kb * 32, lane);
            v16bf b = load_b_frag(pk_ns, kb, nt, 8, lane);
            acc1 = wmma_bf16(a, b, acc1);
        }
        v8f acc2 = zero8();
        for (int kb = 0; kb < 24; kb++) {
            v16bf a = load_a_frag(A2, 768, mt * 16, kb * 32, lane);
            v16bf b = load_b_frag(pk_sym, kb, nt, 8, lane);
            acc2 = wmma_bf16(a, b, acc2);
        }
        int n = nt * 16 + (lane & 15);
        for (int r = 0; r < 8; r++) {
            int m = mt * 16 + r + ((lane >> 4) << 3);
            int row = rowbase + m;
            float v1 = silu_f(acc1[r] + b_ns[n]);
            float v2 = silu_f(acc2[r] + b_sym[n]);
            n_partial[row * 128 + n] = node_f[row * 128 + n] + res_n0[n] * v1 + res_n1[n] * v2;
        }
    }
}

// --------------------------- edge GEMM (the big one) -----------------------
// Per loc: A = edge_info [64 x 320] built in LDS; B = [320 x 192]
// (w_node_edge | w_edge_self). node_edge reduced over nei via ds_add_f32.
#define ESTRIDE 328
__global__ void __launch_bounds__(256)
k_edge(const float* __restrict__ edge_f, const int* __restrict__ nlist,
       const float* __restrict__ sw, const bhalf* __restrict__ node_bf,
       const bhalf* __restrict__ edge_bf, const uint4* __restrict__ pk,
       const float* __restrict__ b_ne, const float* __restrict__ b_es,
       const float* __restrict__ n_partial, const float* __restrict__ res_n2,
       const float* __restrict__ res_e0, float* __restrict__ out_n,
       float* __restrict__ out_e) {
    int loc = blockIdx.x;
    __shared__ __align__(16) bhalf sA[64 * ESTRIDE];
    __shared__ float nacc[128];
    __shared__ int   snl[64];
    int t = threadIdx.x;
    if (t < 128) nacc[t] = 0.f;
    if (t < 64) snl[t] = nlist[loc * 64 + t];
    // pull the 120KB packed-B stream toward the WGP while we stage A
    for (int i = t; i < 960; i += 256)
        __builtin_prefetch((const char*)pk + i * 128, 0, 1);
    __syncthreads();
    for (int ch = t; ch < 1024; ch += 256) {             // node_i (broadcast)
        int m = ch >> 4, c = (ch & 15) * 8;
        cp16(node_bf + loc * 128 + c, &sA[m * ESTRIDE + c]);
    }
    for (int ch = t; ch < 1024; ch += 256) {             // nei_node (gather)
        int m = ch >> 4, c = (ch & 15) * 8;
        cp16(node_bf + snl[m] * 128 + c, &sA[m * ESTRIDE + 128 + c]);
    }
    for (int ch = t; ch < 512; ch += 256) {              // edge_ebd
        int m = ch >> 3, c = (ch & 7) * 8;
        cp16(edge_bf + loc * 4096 + m * 64 + c, &sA[m * ESTRIDE + 256 + c]);
    }
    cp_wait();
    __syncthreads();
    int lane = t & 31, w = t >> 5;
    int mt = w & 3, ntb = (w >> 2) * 6;
    v8f acc[6] = {zero8(), zero8(), zero8(), zero8(), zero8(), zero8()};
    for (int kb = 0; kb < 10; kb++) {
        v16bf a = load_a_frag(sA, ESTRIDE, mt * 16, kb * 32, lane);
        for (int j = 0; j < 6; j++) {
            v16bf b = load_b_frag(pk, kb, ntb + j, 12, lane);
            acc[j] = wmma_bf16(a, b, acc[j]);
        }
    }
    for (int j = 0; j < 6; j++) {
        int n = (ntb + j) * 16 + (lane & 15);
        for (int r = 0; r < 8; r++) {
            int m = mt * 16 + r + ((lane >> 4) << 3);
            if (n < 128) {
                float s = silu_f(acc[j][r] + b_ne[n]);
                atomicAdd(&nacc[n], s * sw[loc * 64 + m]);
            } else {
                int ne = n - 128;
                float s = silu_f(acc[j][r] + b_es[ne]);
                int ei = loc * 4096 + m * 64 + ne;
                out_e[ei] = edge_f[ei] + res_e0[ne] * s;
            }
        }
    }
    __syncthreads();
    if (t < 128)
        out_n[loc * 128 + t] = n_partial[loc * 128 + t] + res_n2[t] * nacc[t] * (1.0f / 64.0f);
}

// --------------------------- angle GEMM (biggest) --------------------------
// Block = (loc, 4 i-values) -> A = angle_info [64 x 288] in LDS,
// B = [288 x 96] (w_ea1 | w_angle_self). eau reduced over j in LDS.
#define ASTRIDE 296
__global__ void __launch_bounds__(256)
k_angle(const float* __restrict__ angle_f, const float* __restrict__ a_sw,
        const bhalf* __restrict__ node_bf, const bhalf* __restrict__ ea_bf,
        const bhalf* __restrict__ angle_bf, const uint4* __restrict__ pk,
        const float* __restrict__ b_ea1, const float* __restrict__ b_as,
        const float* __restrict__ res_a0, bhalf* __restrict__ red_bf,
        float* __restrict__ out_a) {
    int blk = blockIdx.x;
    int loc = blk >> 2;
    int ib = (blk & 3) * 4;
    __shared__ __align__(16) bhalf sA[64 * ASTRIDE];
    __shared__ float racc[256];
    int t = threadIdx.x;
    racc[t] = 0.f;
    for (int ch = t; ch < 64 * 36; ch += 256) {
        int m = ch / 36;
        int c = (ch - m * 36) * 8;
        int il = m >> 4, j = m & 15, i = ib + il;
        const bhalf* src;
        if (c < 32)       src = angle_bf + ((loc * 16 + i) * 16 + j) * 32 + c;
        else if (c < 160) src = node_bf + loc * 128 + (c - 32);
        else if (c < 224) src = ea_bf + (loc * 16 + j) * 64 + (c - 160);  // ea_i[l,i,j]=ea[l,j]
        else              src = ea_bf + (loc * 16 + i) * 64 + (c - 224);  // ea_j[l,i,j]=ea[l,i]
        cp16(src, &sA[m * ASTRIDE + c]);
    }
    cp_wait();
    __syncthreads();
    int lane = t & 31, w = t >> 5;
    int mt = w & 3, ntb = (w >> 2) * 3;
    v8f acc[3] = {zero8(), zero8(), zero8()};
    for (int kb = 0; kb < 9; kb++) {
        v16bf a = load_a_frag(sA, ASTRIDE, mt * 16, kb * 32, lane);
        for (int j = 0; j < 3; j++) {
            v16bf b = load_b_frag(pk, kb, ntb + j, 6, lane);
            acc[j] = wmma_bf16(a, b, acc[j]);
        }
    }
    for (int jj = 0; jj < 3; jj++) {
        int n = (ntb + jj) * 16 + (lane & 15);
        for (int r = 0; r < 8; r++) {
            int m = mt * 16 + r + ((lane >> 4) << 3);
            int il = m >> 4, aj = m & 15, i = ib + il;
            if (n < 64) {
                float s = silu_f(acc[jj][r] + b_ea1[n]);
                float wgt = a_sw[loc * 16 + i] * a_sw[loc * 16 + aj];
                atomicAdd(&racc[il * 64 + n], wgt * s);
            } else {
                int na = n - 64;
                float s = silu_f(acc[jj][r] + b_as[na]);
                int ai = ((loc * 16 + i) * 16 + aj) * 32 + na;
                out_a[ai] = angle_f[ai] + res_a0[na] * s;
            }
        }
    }
    __syncthreads();
    {
        int il = t >> 6, n = t & 63;
        red_bf[(loc * 16 + ib + il) * 64 + n] = (bhalf)(racc[t] * 0.25f); // 1/sqrt(16)
    }
}

// --------------------------- e_angle GEMM ----------------------------------
// padded[m] = (m<16 && amask) ? reduced : edge_ebd; out_e += res_e1*silu(padded@Wea2+b)
#define E2STRIDE 72
__global__ void __launch_bounds__(256)
k_edge2(const bhalf* __restrict__ edge_bf, const unsigned char* __restrict__ amask,
        const bhalf* __restrict__ red_bf, const uint4* __restrict__ pk,
        const float* __restrict__ b_ea2, const float* __restrict__ res_e1,
        float* __restrict__ out_e) {
    int loc = blockIdx.x;
    __shared__ __align__(16) bhalf sA[64 * E2STRIDE];
    int t = threadIdx.x;
    for (int ch = t; ch < 512; ch += 256) {
        int m = ch >> 3, c = (ch & 7) * 8;
        bool use_red = (m < 16) && (amask[loc * 16 + m] != 0);
        const bhalf* src = use_red ? (red_bf + (loc * 16 + m) * 64 + c)
                                   : (edge_bf + loc * 4096 + m * 64 + c);
        cp16(src, &sA[m * E2STRIDE + c]);
    }
    cp_wait();
    __syncthreads();
    int lane = t & 31, w = t >> 5;
    int mt = w & 3, ntb = (w >> 2) * 2;
    v8f acc[2] = {zero8(), zero8()};
    for (int kb = 0; kb < 2; kb++) {
        v16bf a = load_a_frag(sA, E2STRIDE, mt * 16, kb * 32, lane);
        for (int j = 0; j < 2; j++) {
            v16bf b = load_b_frag(pk, kb, ntb + j, 4, lane);
            acc[j] = wmma_bf16(a, b, acc[j]);
        }
    }
    for (int j = 0; j < 2; j++) {
        int n = (ntb + j) * 16 + (lane & 15);
        for (int r = 0; r < 8; r++) {
            int m = mt * 16 + r + ((lane >> 4) << 3);
            float s = silu_f(acc[j][r] + b_ea2[n]);
            int ei = loc * 4096 + m * 64 + n;
            out_e[ei] = out_e[ei] + res_e1[n] * s;
        }
    }
}

// --------------------------- launch ----------------------------------------

extern "C" void kernel_launch(void* const* d_in, const int* in_sizes, int n_in,
                              void* d_out, int out_size, void* d_ws, size_t ws_size,
                              hipStream_t stream) {
    const float* node_f  = (const float*)d_in[0];
    const float* edge_f  = (const float*)d_in[1];
    const float* h2      = (const float*)d_in[2];
    const float* angle_f = (const float*)d_in[3];
    const int*   nlist   = (const int*)d_in[4];
    const unsigned char* nmask = (const unsigned char*)d_in[5];
    const float* sw      = (const float*)d_in[6];
    const unsigned char* amask = (const unsigned char*)d_in[7];
    const float* a_sw    = (const float*)d_in[8];
    const float* w_ns  = (const float*)d_in[9];  const float* b_ns  = (const float*)d_in[10];
    const float* w_sym = (const float*)d_in[11]; const float* b_sym = (const float*)d_in[12];
    const float* w_ne  = (const float*)d_in[13]; const float* b_ne  = (const float*)d_in[14];
    const float* w_es  = (const float*)d_in[15]; const float* b_es  = (const float*)d_in[16];
    const float* w_ea1 = (const float*)d_in[17]; const float* b_ea1 = (const float*)d_in[18];
    const float* w_ea2 = (const float*)d_in[19]; const float* b_ea2 = (const float*)d_in[20];
    const float* w_as  = (const float*)d_in[21]; const float* b_as  = (const float*)d_in[22];
    const float* res_n0 = (const float*)d_in[23];
    const float* res_n1 = (const float*)d_in[24];
    const float* res_n2 = (const float*)d_in[25];
    const float* res_e0 = (const float*)d_in[26];
    const float* res_e1 = (const float*)d_in[27];
    const float* res_a0 = (const float*)d_in[28];

    float* out_n = (float*)d_out;                       // [1024,128]
    float* out_e = out_n + 1024 * 128;                  // [1024,64,64]
    float* out_a = out_e + 1024 * 64 * 64;              // [1024,16,16,32]

    // workspace carve-out (256B aligned)
    char* ws = (char*)d_ws;
    size_t off = 0;
    auto carve = [&](size_t bytes) {
        void* p = ws + off;
        off += (bytes + 255) & ~(size_t)255;
        return p;
    };
    bhalf* node_bf  = (bhalf*)carve(1024 * 128 * 2);
    bhalf* edge_bf  = (bhalf*)carve(1024 * 64 * 64 * 2);
    bhalf* ea_bf    = (bhalf*)carve(1024 * 16 * 64 * 2);
    bhalf* angle_bf = (bhalf*)carve(1024 * 16 * 16 * 32 * 2);
    bhalf* sym_bf   = (bhalf*)carve(1024 * 768 * 2);
    float* n_part   = (float*)carve(1024 * 128 * 4);
    bhalf* red_bf   = (bhalf*)carve(1024 * 16 * 64 * 2);
    unsigned* pk_ns   = (unsigned*)carve(128 * 128 / 2 * 4);
    unsigned* pk_sym  = (unsigned*)carve(768 * 128 / 2 * 4);
    unsigned* pk_nees = (unsigned*)carve(320 * 192 / 2 * 4);
    unsigned* pk_ea   = (unsigned*)carve(288 * 96 / 2 * 4);
    unsigned* pk_ea2  = (unsigned*)carve(64 * 64 / 2 * 4);

    // 1) bf16 conversion + weight packing
    k_cvt<<<512, 256, 0, stream>>>(node_f, node_bf, 1024 * 128);
    k_cvt<<<512, 256, 0, stream>>>(edge_f, edge_bf, 1024 * 64 * 64);
    k_cvt<<<512, 256, 0, stream>>>(angle_f, angle_bf, 1024 * 16 * 16 * 32);
    k_ea<<<256, 256, 0, stream>>>(edge_f, amask, ea_bf);
    k_pack<<<64, 256, 0, stream>>>(w_ns, 128, nullptr, 0, 128, pk_ns);
    k_pack<<<64, 256, 0, stream>>>(w_sym, 128, nullptr, 0, 768, pk_sym);
    k_pack<<<64, 256, 0, stream>>>(w_ne, 128, w_es, 64, 320, pk_nees);
    k_pack<<<64, 256, 0, stream>>>(w_ea1, 64, w_as, 32, 288, pk_ea);
    k_pack<<<64, 256, 0, stream>>>(w_ea2, 64, nullptr, 0, 64, pk_ea2);

    // 2) sym vectors (VALU, tiny)
    k_sym<<<1024, 128, 0, stream>>>(node_f, edge_f, h2, nlist, nmask, sw, sym_bf);

    // 3) node GEMMs -> n_partial
    k_node<<<16, 256, 0, stream>>>(node_f, node_bf, sym_bf, (const uint4*)pk_ns,
                                   (const uint4*)pk_sym, b_ns, b_sym, res_n0, res_n1, n_part);

    // 4) edge_info GEMM: edge_self + node_edge reduction + final n_updated
    k_edge<<<1024, 256, 0, stream>>>(edge_f, nlist, sw, node_bf, edge_bf,
                                     (const uint4*)pk_nees, b_ne, b_es, n_part,
                                     res_n2, res_e0, out_n, out_e);

    // 5) angle_info GEMM: eau (j-reduced) + a_updated
    k_angle<<<4096, 256, 0, stream>>>(angle_f, a_sw, node_bf, ea_bf, angle_bf,
                                      (const uint4*)pk_ea, b_ea1, b_as, res_a0,
                                      red_bf, out_a);

    // 6) e_angle GEMM: finalize e_updated
    k_edge2<<<1024, 256, 0, stream>>>(edge_bf, amask, red_bf, (const uint4*)pk_ea2,
                                      b_ea2, res_e1, out_e);
}
